// SiameseGAT_4209067950752
// MI455X (gfx1250) — matrix operand order repaired and checked by reading, not compiled, not run on hardware
//
#include <hip/hip_runtime.h>
#include <hip/hip_bf16.h>

// ---------------------------------------------------------------------------
// Siamese GAT forward for MI455X (gfx1250, wave32).
// The six [40000,128]@[128,128] node-feature GEMMs use v_wmma_f32_16x16x32_f16
// with the ISA-documented per-lane fragment layouts. Everything else is
// bandwidth/atomic bound edge + segment-softmax work.
// ---------------------------------------------------------------------------

#define Nn 40000
#define Ee 512000
#define HC 128
#define Hh 4
#define Gg 64
#define FCk 48

typedef __attribute__((ext_vector_type(16))) _Float16 v16h;
typedef __attribute__((ext_vector_type(8)))  float    v8f;

// ---------------------------- utility kernels ------------------------------

__global__ void fill_kernel(float* __restrict__ p, long long n, float v) {
  long long i = (long long)blockIdx.x * blockDim.x + threadIdx.x;
  if (i < n) p[i] = v;
}

__device__ __forceinline__ void atomicMaxFloat(float* addr, float val) {
  // signed-max for val>=0, unsigned-min for val<0; works with -inf init.
  if (val >= 0.f) atomicMax((int*)addr, __float_as_int(val));
  else            atomicMin((unsigned int*)addr, __float_as_uint(val));
}

// ------------------------- WMMA GEMM  H = X @ W ----------------------------
// X: [nrows,128] f32 row-major, W: [128,128] f32 row-major, H: [nrows,128] f32.
// One wave -> 16 rows x 128 cols. W staged in LDS as f16 (32 KB).

__global__ __launch_bounds__(128) void gemm128_wmma(
    const float* __restrict__ X, const float* __restrict__ W,
    float* __restrict__ Hout, int nrows) {
  __shared__ _Float16 Wl[128 * 128];
  for (int i = threadIdx.x; i < 128 * 128; i += 128)
    Wl[i] = (_Float16)W[i];
  __syncthreads();

  const int wave = threadIdx.x >> 5;
  const int lane = threadIdx.x & 31;
  const int lp   = lane & 15;
  const int half = lane >> 4;
  const int r0   = (blockIdx.x * 4 + wave) * 16;
  if (r0 >= nrows) return;  // never taken: nrows % 64 == 0

  v8f acc[8] = {};
  const float* xrow = X + (size_t)(r0 + lp) * 128;

  for (int kb = 0; kb < 4; ++kb) {
    const int k0 = kb * 32;
    // A fragment: 16-bit A 16x32 layout (ISA 7.12.2)
    v16h a;
#pragma unroll
    for (int e = 0; e < 16; ++e) {
      int k = k0 + ((e >= 8) ? 16 : 0) + half * 8 + ((e >> 1) & 3) * 2 + (e & 1);
      a[e] = (_Float16)xrow[k];
    }
#pragma unroll
    for (int t = 0; t < 8; ++t) {
      const int c0 = t * 16;
      // B fragment: 32x16, K = (e>=8?16:0) + half*8 + (e&7), n = lp
      v16h b;
#pragma unroll
      for (int e = 0; e < 16; ++e)
        b[e] = Wl[(k0 + ((e >= 8) ? 16 : 0) + half * 8 + (e & 7)) * 128 + c0 + lp];
      acc[t] = __builtin_amdgcn_wmma_f32_16x16x32_f16(
          false, a, false, b, (short)0, acc[t], false, false);
    }
  }
  // D layout: VGPR v -> row v + 8*half, col lp
#pragma unroll
  for (int t = 0; t < 8; ++t)
#pragma unroll
    for (int v = 0; v < 8; ++v)
      Hout[(size_t)(r0 + v + 8 * half) * 128 + t * 16 + lp] = acc[t][v];
}

// ------------------- per-node per-head attention scores --------------------
// block = node, warp = head: es[n,h] = h[n,h,:].a_s[h,:], ed likewise.

__global__ __launch_bounds__(128) void scores_kernel(
    const float* __restrict__ h, const float* __restrict__ as_,
    const float* __restrict__ ad_, float* __restrict__ es,
    float* __restrict__ ed) {
  const int node = blockIdx.x;
  const int head = threadIdx.x >> 5;
  const int lane = threadIdx.x & 31;
  const float v = h[(size_t)node * 128 + head * 32 + lane];
  float s = v * as_[head * 32 + lane];
  float d = v * ad_[head * 32 + lane];
  for (int off = 16; off; off >>= 1) {
    s += __shfl_down(s, off, 32);
    d += __shfl_down(d, off, 32);
  }
  if (lane == 0) { es[node * 4 + head] = s; ed[node * 4 + head] = d; }
}

// ------------------------------ edge passes --------------------------------

__global__ void edge_max_kernel(const int* __restrict__ ei,
                                const float* __restrict__ es,
                                const float* __restrict__ ed,
                                float* __restrict__ m) {
  int idx = blockIdx.x * blockDim.x + threadIdx.x;
  const int total = (Ee + Nn) * Hh;
  if (idx >= total) return;
  int j = idx >> 2, hh = idx & 3, s, d;
  if (j < Ee) { s = ei[j]; d = ei[Ee + j]; } else { s = d = j - Ee; }
  float e = es[s * 4 + hh] + ed[d * 4 + hh];
  e = e > 0.f ? e : 0.2f * e;  // leaky_relu(0.2)
  atomicMaxFloat(&m[d * 4 + hh], e);
}

__global__ void edge_exp_kernel(const int* __restrict__ ei,
                                const float* __restrict__ es,
                                const float* __restrict__ ed,
                                const float* __restrict__ m,
                                float* __restrict__ alphab,
                                float* __restrict__ den) {
  int idx = blockIdx.x * blockDim.x + threadIdx.x;
  const int total = (Ee + Nn) * Hh;
  if (idx >= total) return;
  int j = idx >> 2, hh = idx & 3, s, d;
  if (j < Ee) { s = ei[j]; d = ei[Ee + j]; } else { s = d = j - Ee; }
  float e = es[s * 4 + hh] + ed[d * 4 + hh];
  e = e > 0.f ? e : 0.2f * e;
  float ex = expf(e - m[d * 4 + hh]);
  alphab[j * 4 + hh] = ex;
  atomicAdd(&den[d * 4 + hh], ex);
}

__global__ void edge_scatter_kernel(const int* __restrict__ ei,
                                    const float* __restrict__ h,
                                    const float* __restrict__ alphab,
                                    const float* __restrict__ den,
                                    float* __restrict__ out) {
  long long idx = (long long)blockIdx.x * blockDim.x + threadIdx.x;
  const long long total = (long long)(Ee + Nn) * HC;
  if (idx >= total) return;
  int j = (int)(idx >> 7), c = (int)(idx & 127), hh = c >> 5, s, d;
  if (j < Ee) { s = ei[j]; d = ei[Ee + j]; } else { s = d = j - Ee; }
  float a = alphab[j * 4 + hh] / den[d * 4 + hh];
  atomicAdd(&out[(size_t)d * 128 + c], h[(size_t)s * 128 + c] * a);
}

// ------------------------------ BatchNorm ----------------------------------
// Note: the GAT bias feeds straight into training-mode BN and
// cancels exactly, so it is not added before stats/apply.

__global__ __launch_bounds__(256) void bn_stats_kernel(
    const float* __restrict__ x, int n, float* __restrict__ mean,
    float* __restrict__ var) {
  const int c = blockIdx.x;  // 128 channels
  __shared__ float s1[256], s2[256];
  float a = 0.f, b = 0.f;
  for (int i = threadIdx.x; i < n; i += 256) {
    float v = x[(size_t)i * 128 + c];
    a += v; b += v * v;
  }
  s1[threadIdx.x] = a; s2[threadIdx.x] = b;
  __syncthreads();
  for (int off = 128; off; off >>= 1) {
    if (threadIdx.x < off) {
      s1[threadIdx.x] += s1[threadIdx.x + off];
      s2[threadIdx.x] += s2[threadIdx.x + off];
    }
    __syncthreads();
  }
  if (threadIdx.x == 0) {
    float mu = s1[0] / n;
    mean[c] = mu;
    var[c] = s2[0] / n - mu * mu;  // biased
  }
}

__global__ void bn_relu_kernel(float* __restrict__ x,
                               const float* __restrict__ mean,
                               const float* __restrict__ var,
                               const float* __restrict__ g,
                               const float* __restrict__ be, long long total) {
  long long i = (long long)blockIdx.x * blockDim.x + threadIdx.x;
  if (i >= total) return;
  int c = (int)(i & 127);
  float y = g[c] * (x[i] - mean[c]) * rsqrtf(var[c] + 1e-5f) + be[c];
  x[i] = y > 0.f ? y : 0.f;
}

// ----------------------------- attention pool ------------------------------

__global__ __launch_bounds__(128) void pool_score_kernel(
    const float* __restrict__ x, const float* __restrict__ att_w,
    const float* __restrict__ att_b, const int* __restrict__ batch,
    float* __restrict__ sc, float* __restrict__ pm) {
  const int warp = threadIdx.x >> 5, lane = threadIdx.x & 31;
  const int node = blockIdx.x * 4 + warp;
  if (node >= Nn) return;
  float s = 0.f;
  for (int c = lane; c < 128; c += 32) s += x[(size_t)node * 128 + c] * att_w[c];
  for (int off = 16; off; off >>= 1) s += __shfl_down(s, off, 32);
  if (lane == 0) {
    s += att_b[0];
    sc[node] = s;
    atomicMaxFloat(&pm[batch[node]], s);
  }
}

__global__ void pool_exp_kernel(const float* __restrict__ sc,
                                const int* __restrict__ batch,
                                const float* __restrict__ pm,
                                float* __restrict__ pex,
                                float* __restrict__ pden) {
  int n = blockIdx.x * blockDim.x + threadIdx.x;
  if (n >= Nn) return;
  int b = batch[n];
  float ex = expf(sc[n] - pm[b]);
  pex[n] = ex;
  atomicAdd(&pden[b], ex);
}

__global__ void pool_scatter_kernel(const float* __restrict__ x,
                                    const int* __restrict__ batch,
                                    const float* __restrict__ pex,
                                    const float* __restrict__ pden,
                                    float* __restrict__ p) {
  long long idx = (long long)blockIdx.x * blockDim.x + threadIdx.x;
  const long long total = (long long)Nn * HC;
  if (idx >= total) return;
  int n = (int)(idx >> 7), c = (int)(idx & 127);
  int b = batch[n];
  atomicAdd(&p[b * 128 + c], x[idx] * pex[n] / pden[b]);
}

// --------------------------- tiny dense layers -----------------------------

__global__ void fc_kernel(const float* __restrict__ X,
                          const float* __restrict__ Wm,
                          const float* __restrict__ bias,
                          float* __restrict__ Y, int rows, int kin, int kout,
                          int out_ld, int out_off) {
  int idx = blockIdx.x * blockDim.x + threadIdx.x;
  if (idx >= rows * kout) return;
  int g = idx / kout, j = idx % kout;
  float s = bias ? bias[j] : 0.f;
  for (int i = 0; i < kin; ++i) s += X[g * kin + i] * Wm[i * kout + j];
  Y[g * out_ld + out_off + j] = s;
}

__global__ void bn_small_relu_kernel(float* __restrict__ X,
                                     const float* __restrict__ g,
                                     const float* __restrict__ be, int rows,
                                     int cols) {
  int j = threadIdx.x;
  if (j >= cols) return;
  float s = 0.f, q = 0.f;
  for (int r = 0; r < rows; ++r) {
    float v = X[r * cols + j];
    s += v; q += v * v;
  }
  float mu = s / rows;
  float inv = rsqrtf(q / rows - mu * mu + 1e-5f);
  for (int r = 0; r < rows; ++r) {
    float v = g[j] * (X[r * cols + j] - mu) * inv + be[j];
    X[r * cols + j] = v > 0.f ? v : 0.f;
  }
}

__global__ void final_kernel(const float* __restrict__ o1,
                             const float* __restrict__ w,
                             const float* __restrict__ b,
                             float* __restrict__ out, int kin) {
  int g = blockIdx.x * blockDim.x + threadIdx.x;
  if (g >= Gg) return;
  float s = b[0];
  for (int i = 0; i < kin; ++i) s += o1[g * kin + i] * w[i];
  out[g] = 1.f / (1.f + expf(-s));
}

// ------------------------------- host side ---------------------------------

static inline int cdiv(long long a, int b) { return (int)((a + b - 1) / b); }

extern "C" void kernel_launch(void* const* d_in, const int* in_sizes, int n_in,
                              void* d_out, int out_size, void* d_ws,
                              size_t ws_size, hipStream_t stream) {
  (void)in_sizes; (void)n_in; (void)out_size; (void)ws_size;
  const float NEG_INF = -__builtin_huge_valf();

  // ---- input map (setup_inputs dict order) ----
  const float* Wl[3]  = {(const float*)d_in[6], (const float*)d_in[10], (const float*)d_in[14]};
  const float* asl[3] = {(const float*)d_in[7], (const float*)d_in[11], (const float*)d_in[15]};
  const float* adl[3] = {(const float*)d_in[8], (const float*)d_in[12], (const float*)d_in[16]};
  const float* gl[3]  = {(const float*)d_in[18], (const float*)d_in[19], (const float*)d_in[20]};
  const float* bel[3] = {(const float*)d_in[23], (const float*)d_in[24], (const float*)d_in[25]};
  const float* gbi   = (const float*)d_in[21];
  const float* gf1   = (const float*)d_in[22];
  const float* bbi   = (const float*)d_in[26];
  const float* bf1   = (const float*)d_in[27];
  const float* att_w = (const float*)d_in[28];
  const float* att_b = (const float*)d_in[29];
  const float* fci_w = (const float*)d_in[30];
  const float* fci_b = (const float*)d_in[31];
  const float* fcw   = (const float*)d_in[32];
  const float* fcb   = (const float*)d_in[33];
  const float* fc1w  = (const float*)d_in[34];
  const float* fc1b  = (const float*)d_in[35];
  const float* fc2w  = (const float*)d_in[36];
  const float* fc2b  = (const float*)d_in[37];

  // ---- workspace layout (floats) ----
  float* w = (float*)d_ws;
  float* A      = w;               w += (size_t)Nn * HC;      // gemm output h
  float* B      = w;               w += (size_t)Nn * HC;      // scatter/act buffer
  float* es     = w;               w += (size_t)Nn * Hh;
  float* ed     = w;               w += (size_t)Nn * Hh;
  float* mmax   = w;               w += (size_t)Nn * Hh;
  float* den    = w;               w += (size_t)Nn * Hh;
  float* alphab = w;               w += (size_t)(Ee + Nn) * Hh;
  float* sc     = w;               w += Nn;
  float* pex    = w;               w += Nn;
  float* pm     = w;               w += Gg;
  float* pden   = w;               w += Gg;
  float* pool   = w;               w += Gg * HC;
  float* zbuf   = w;               w += Gg * FCk;
  float* ccbuf  = w;               w += Gg * 24;
  float* o1buf  = w;               w += Gg * 12;
  float* meanb  = w;               w += HC;
  float* varb   = w;               w += HC;

  const int etot4   = (Ee + Nn) * Hh;
  const long long etotc = (long long)(Ee + Nn) * HC;
  const long long nhc   = (long long)Nn * HC;

  for (int br = 0; br < 2; ++br) {
    const float* xin   = (const float*)d_in[br];       // x1 / x2
    const int*   ei    = (const int*)d_in[2 + br];     // edge_index
    const int*   batch = (const int*)d_in[4 + br];

    const float* Xcur = xin;
    for (int l = 0; l < 3; ++l) {
      // h = Xcur @ W[l]   (WMMA)
      gemm128_wmma<<<Nn / 64, 128, 0, stream>>>(Xcur, Wl[l], A, Nn);
      // per-head attention scores
      scores_kernel<<<Nn, 128, 0, stream>>>(A, asl[l], adl[l], es, ed);
      // clear segment buffers + output accumulator
      fill_kernel<<<cdiv((long long)Nn * Hh, 256), 256, 0, stream>>>(mmax, (long long)Nn * Hh, NEG_INF);
      fill_kernel<<<cdiv((long long)Nn * Hh, 256), 256, 0, stream>>>(den, (long long)Nn * Hh, 0.f);
      fill_kernel<<<cdiv(nhc, 256), 256, 0, stream>>>(B, nhc, 0.f);
      // segment softmax over incoming edges (+ self loops)
      edge_max_kernel<<<cdiv(etot4, 256), 256, 0, stream>>>(ei, es, ed, mmax);
      edge_exp_kernel<<<cdiv(etot4, 256), 256, 0, stream>>>(ei, es, ed, mmax, alphab, den);
      edge_scatter_kernel<<<cdiv(etotc, 256), 256, 0, stream>>>(ei, A, alphab, den, B);
      // BN (training stats) + ReLU in place (GAT bias cancels in BN)
      bn_stats_kernel<<<HC, 256, 0, stream>>>(B, Nn, meanb, varb);
      bn_relu_kernel<<<cdiv(nhc, 256), 256, 0, stream>>>(B, meanb, varb, gl[l], bel[l], nhc);
      Xcur = B;
    }

    // ---- attention pool over graphs ----
    fill_kernel<<<1, 256, 0, stream>>>(pm, Gg, NEG_INF);
    fill_kernel<<<1, 256, 0, stream>>>(pden, Gg, 0.f);
    fill_kernel<<<cdiv(Gg * HC, 256), 256, 0, stream>>>(pool, Gg * HC, 0.f);
    pool_score_kernel<<<Nn / 4, 128, 0, stream>>>(B, att_w, att_b, batch, sc, pm);
    pool_exp_kernel<<<cdiv(Nn, 256), 256, 0, stream>>>(sc, batch, pm, pex, pden);
    pool_scatter_kernel<<<cdiv(nhc, 256), 256, 0, stream>>>(B, batch, pex, pden, pool);

    // ---- head: relu(bn(p @ fci_w + fci_b)) @ fcw + fcb -> cc[:, br*12:] ----
    fc_kernel<<<cdiv(Gg * FCk, 256), 256, 0, stream>>>(pool, fci_w, fci_b, zbuf,
                                                       Gg, HC, FCk, FCk, 0);
    bn_small_relu_kernel<<<1, 64, 0, stream>>>(zbuf, gbi, bbi, Gg, FCk);
    fc_kernel<<<cdiv(Gg * 12, 256), 256, 0, stream>>>(zbuf, fcw, fcb, ccbuf,
                                                      Gg, FCk, 12, 24, br * 12);
  }

  // ---- final: relu(bn(cc @ fc1w + fc1b)) @ fc2w + fc2b -> sigmoid ----
  fc_kernel<<<cdiv(Gg * 12, 256), 256, 0, stream>>>(ccbuf, fc1w, fc1b, o1buf,
                                                    Gg, 24, 12, 12, 0);
  bn_small_relu_kernel<<<1, 32, 0, stream>>>(o1buf, gf1, bf1, Gg, 12);
  final_kernel<<<1, 64, 0, stream>>>(o1buf, fc2w, fc2b, (float*)d_out, 12);
}
